// GraphAttentionLayer_82746839925327
// MI455X (gfx1250) — matrix operand order repaired and checked by reading, not compiled, not run on hardware
//
#include <hip/hip_runtime.h>
#include <hip/hip_bf16.h>
#include <hip/hip_fp16.h>

// GraphAttentionLayer for MI455X (gfx1250, wave32).
// Fused flash-style column-softmax attention; main matmul on v_wmma_f32_16x16x32_f16.

constexpr int NV    = 8192;  // number of nodes
constexpr int IN_F  = 48;
constexpr int OUT_F = 64;
constexpr int NCLS  = 16;
#define ALPHA_LR 0.2f

typedef _Float16 v16h __attribute__((ext_vector_type(16)));
typedef float    v8f  __attribute__((ext_vector_type(8)));

__device__ __forceinline__ unsigned flipf(float f) {
  unsigned u = __float_as_uint(f);
  return (u & 0x80000000u) ? ~u : (u | 0x80000000u);
}
__device__ __forceinline__ float unflipf(unsigned u) {
  u = (u & 0x80000000u) ? (u ^ 0x80000000u) : ~u;
  return __uint_as_float(u);
}

// ---------------------------------------------------------------------------
// 1) h = input @ W (f32 exact), h16T[f][i] = (f16)h[i][f], s1 = h@a1, s2 = h@a2,
//    cls[i] = clip(labels[i]-1, 0, 15).  One block (64 threads) per row i.
// ---------------------------------------------------------------------------
__global__ void prep_kernel(const float* __restrict__ inp, const float* __restrict__ W,
                            const float* __restrict__ a, const int* __restrict__ labels,
                            _Float16* __restrict__ h16T, float* __restrict__ s1,
                            float* __restrict__ s2, int* __restrict__ cls) {
  int i = blockIdx.x;
  int f = threadIdx.x;
  const float* row = inp + (size_t)i * IN_F;
  float acc = 0.f;
#pragma unroll
  for (int k = 0; k < IN_F; ++k) acc = fmaf(row[k], W[k * OUT_F + f], acc);
  h16T[(size_t)f * NV + i] = (_Float16)acc;

  __shared__ float r1[OUT_F], r2[OUT_F];
  r1[f] = acc * a[f];
  r2[f] = acc * a[OUT_F + f];
  __syncthreads();
  for (int s = OUT_F / 2; s > 0; s >>= 1) {
    if (f < s) { r1[f] += r1[f + s]; r2[f] += r2[f + s]; }
    __syncthreads();
  }
  if (f == 0) {
    s1[i] = r1[0];
    s2[i] = r2[0];
    int l = labels[i] - 1;
    cls[i] = l < 0 ? 0 : (l > NCLS - 1 ? NCLS - 1 : l);
  }
}

// ---------------------------------------------------------------------------
// 2) maxs1[c] = max_{i: cls[i]==c} s1[i]  (flip-int atomicMax in LDS)
// ---------------------------------------------------------------------------
__global__ void classmax_kernel(const float* __restrict__ s1, const int* __restrict__ cls,
                                float* __restrict__ maxs1) {
  __shared__ unsigned smax[NCLS];
  int t = threadIdx.x;
  if (t < NCLS) smax[t] = flipf(-__builtin_inff());
  __syncthreads();
  for (int i = t; i < NV; i += blockDim.x)
    atomicMax(&smax[cls[i]], flipf(s1[i]));
  __syncthreads();
  if (t < NCLS) maxs1[t] = unflipf(smax[t]);
}

// ---------------------------------------------------------------------------
// 3) colmax[j] = max_c [ LR(maxs1[c]+s2[j]) + alpha*M[c, cls[j]] ]  (LR monotonic)
//    Also zero colsum.
// ---------------------------------------------------------------------------
__global__ void colmax_kernel(const float* __restrict__ s2, const int* __restrict__ cls,
                              const float* __restrict__ maxs1, const float* __restrict__ Mint,
                              const float* __restrict__ alpha_p,
                              float* __restrict__ colmax, float* __restrict__ colsum) {
  int j = blockIdx.x * blockDim.x + threadIdx.x;
  if (j >= NV) return;
  float alpha = *alpha_p;
  float s2j = s2[j];
  int cj = cls[j];
  float cm = -__builtin_inff();
#pragma unroll
  for (int c = 0; c < NCLS; ++c) {
    float ee = maxs1[c] + s2j;
    ee = ee > 0.f ? ee : ALPHA_LR * ee;
    ee += alpha * Mint[c * NCLS + cj];
    cm = fmaxf(cm, ee);
  }
  colmax[j] = cm;
  colsum[j] = 0.f;
}

// ---------------------------------------------------------------------------
// 4) colsum[j] += sum_{i in block range} exp(e[i,j] - colmax[j])
//    grid = (NV/256 j-blocks, 8 i-blocks); s1/cls tiles staged in LDS.
// ---------------------------------------------------------------------------
__global__ void colsum_kernel(const float* __restrict__ s1, const float* __restrict__ s2,
                              const int* __restrict__ cls, const float* __restrict__ Mint,
                              const float* __restrict__ alpha_p, const float* __restrict__ colmax,
                              float* __restrict__ colsum) {
  __shared__ float s1s[256];
  __shared__ int   clss[256];
  __shared__ float biasS[NCLS * NCLS];
  int t = threadIdx.x;
  float alpha = *alpha_p;
  if (t < NCLS * NCLS) biasS[t] = alpha * Mint[t];

  int j = blockIdx.x * 256 + t;
  float s2j = s2[j];
  float cm  = colmax[j];
  int   cj  = cls[j];
  float sum = 0.f;
  int ibase = blockIdx.y * 1024;
  for (int tile = 0; tile < 4; ++tile) {
    int ib = ibase + tile * 256;
    __syncthreads();
    s1s[t]  = s1[ib + t];
    clss[t] = cls[ib + t];
    __syncthreads();
    for (int u = 0; u < 256; ++u) {
      float ee = s1s[u] + s2j;
      ee = ee > 0.f ? ee : ALPHA_LR * ee;
      ee += biasS[clss[u] * NCLS + cj];
      sum += __expf(ee - cm);
    }
  }
  atomicAdd(&colsum[j], sum);
}

// ---------------------------------------------------------------------------
// 5) jdata[j] = { s2[j], colmax[j], 1/colsum[j], bits(cls[j]) }
// ---------------------------------------------------------------------------
__global__ void jdata_kernel(const float* __restrict__ s2, const float* __restrict__ colmax,
                             const float* __restrict__ colsum, const int* __restrict__ cls,
                             float4* __restrict__ jdata) {
  int j = blockIdx.x * blockDim.x + threadIdx.x;
  if (j >= NV) return;
  jdata[j] = make_float4(s2[j], colmax[j], 1.0f / colsum[j], __int_as_float(cls[j]));
}

// ---------------------------------------------------------------------------
// 6) out = elu(attention @ h) via v_wmma_f32_16x16x32_f16.
//    Block = 4 waves, one 16-row output tile; each wave covers NV/4 of j.
//    A fragment (attention weights, f16) built in contiguous k order (same as B):
//      lane L: M = L%16, element e -> K = (L/16)*16 + e
//    B fragment loaded contiguous from h16T in the same contiguous k order as A:
//      lane L: N = L%16, element e -> K = (L/16)*16 + e
//    C/D: element r -> M = (L/16)*8 + r, N = L%16.
// ---------------------------------------------------------------------------
__global__ void __launch_bounds__(128) out_kernel(
    const float4* __restrict__ jdata, const float* __restrict__ s1,
    const int* __restrict__ cls, const _Float16* __restrict__ h16T,
    const float* __restrict__ Mint, const float* __restrict__ alpha_p,
    float* __restrict__ out) {
  __shared__ float biasS[NCLS * NCLS];
  __shared__ float accS[4][16 * OUT_F];

  int t    = threadIdx.x;
  int wave = t >> 5;
  int lane = t & 31;
  int hi   = (lane >> 4) & 1;
  int ln   = lane & 15;

  float alpha = *alpha_p;
  if (t < NCLS * NCLS) biasS[t] = alpha * Mint[t];
  __syncthreads();

  int i0 = blockIdx.x * 16;
  float s1i = s1[i0 + ln];               // row m = ln of the tile
  int   ci  = cls[i0 + ln];
  const float* biasrow = &biasS[ci * NCLS];

  v8f C0 = {}, C1 = {}, C2 = {}, C3 = {};
  int jbase = wave * (NV / 4);
  for (int chunk = 0; chunk < (NV / 4) / 32; ++chunk) {
    int j0 = jbase + chunk * 32;
    // Build A = attention weight tile w[16 x 32] in registers (each exp once).
    v16h A;
#pragma unroll
    for (int e = 0; e < 16; ++e) {
      int j = j0 + (hi << 4) + e;
      float4 jd = jdata[j];
      int cj = __float_as_int(jd.w);
      float ee = s1i + jd.x;
      ee = ee > 0.f ? ee : ALPHA_LR * ee;
      ee += biasrow[cj];
      float w = __expf(ee - jd.y) * jd.z;
      A[e] = (_Float16)w;
    }
    // B tiles: h[j0..j0+31, f-tile] from transposed f16 h; contiguous 32B per lane.
    const _Float16* bp = h16T + (size_t)ln * NV + j0 + hi * 16;
    v16h B0 = *(const v16h*)(bp);
    v16h B1 = *(const v16h*)(bp + 16 * (size_t)NV);
    v16h B2 = *(const v16h*)(bp + 32 * (size_t)NV);
    v16h B3 = *(const v16h*)(bp + 48 * (size_t)NV);
    C0 = __builtin_amdgcn_wmma_f32_16x16x32_f16(false, A, false, B0, (short)0, C0, false, false);
    C1 = __builtin_amdgcn_wmma_f32_16x16x32_f16(false, A, false, B1, (short)0, C1, false, false);
    C2 = __builtin_amdgcn_wmma_f32_16x16x32_f16(false, A, false, B2, (short)0, C2, false, false);
    C3 = __builtin_amdgcn_wmma_f32_16x16x32_f16(false, A, false, B3, (short)0, C3, false, false);
  }

  // Spill per-wave partials to LDS in logical (M, N) order.
#pragma unroll
  for (int r = 0; r < 8; ++r) {
    int M = hi * 8 + r;
    accS[wave][M * OUT_F +  0 + ln] = C0[r];
    accS[wave][M * OUT_F + 16 + ln] = C1[r];
    accS[wave][M * OUT_F + 32 + ln] = C2[r];
    accS[wave][M * OUT_F + 48 + ln] = C3[r];
  }
  __syncthreads();

  // Combine 4 j-range partials, apply ELU, store.
  for (int q = t; q < 16 * OUT_F; q += 128) {
    float v = accS[0][q] + accS[1][q] + accS[2][q] + accS[3][q];
    v = v > 0.f ? v : (__expf(v) - 1.0f);
    out[(size_t)(i0 + (q >> 6)) * OUT_F + (q & 63)] = v;
  }
}

// ---------------------------------------------------------------------------
extern "C" void kernel_launch(void* const* d_in, const int* in_sizes, int n_in,
                              void* d_out, int out_size, void* d_ws, size_t ws_size,
                              hipStream_t stream) {
  (void)in_sizes; (void)n_in; (void)out_size; (void)ws_size;
  const float* inp    = (const float*)d_in[0];
  // d_in[1] = adj : all-ones, unused by the reference computation.
  const float* W      = (const float*)d_in[2];
  const float* a      = (const float*)d_in[3];
  const float* Mint   = (const float*)d_in[4];
  const float* alphap = (const float*)d_in[5];
  const int*   labels = (const int*)d_in[6];
  float* out = (float*)d_out;

  // Workspace carve-out (256B aligned slices).
  char* p = (char*)d_ws;
  auto take = [&p](size_t sz) { char* q = p; p += (sz + 255) & ~(size_t)255; return q; };
  _Float16* h16T  = (_Float16*)take((size_t)OUT_F * NV * sizeof(_Float16)); // 1 MB
  float*    s1    = (float*)take(NV * sizeof(float));
  float*    s2    = (float*)take(NV * sizeof(float));
  int*      cls   = (int*)take(NV * sizeof(int));
  float*    maxs1 = (float*)take(NCLS * sizeof(float));
  float*    colmax= (float*)take(NV * sizeof(float));
  float*    colsum= (float*)take(NV * sizeof(float));
  float4*   jdata = (float4*)take(NV * sizeof(float4));                     // 128 KB

  prep_kernel<<<NV, OUT_F, 0, stream>>>(inp, W, a, labels, h16T, s1, s2, cls);
  classmax_kernel<<<1, 256, 0, stream>>>(s1, cls, maxs1);
  colmax_kernel<<<NV / 256, 256, 0, stream>>>(s2, cls, maxs1, Mint, alphap, colmax, colsum);
  colsum_kernel<<<dim3(NV / 256, 8), 256, 0, stream>>>(s1, s2, cls, Mint, alphap, colmax, colsum);
  jdata_kernel<<<NV / 256, 256, 0, stream>>>(s2, colmax, colsum, cls, jdata);
  out_kernel<<<NV / 16, 128, 0, stream>>>(jdata, s1, cls, h16T, Mint, alphap, out);
}
